// Experts_16174846837178
// MI455X (gfx1250) — matrix operand order repaired and checked by reading, not compiled, not run on hardware
//
#include <hip/hip_runtime.h>
#include <stdint.h>

#define NE   8      // experts
#define CAP  4096   // tokens per expert (M)
#define DM   1024   // d_model
#define FF   4096   // d_ff

typedef __attribute__((ext_vector_type(16))) __bf16 v16bf;
typedef __attribute__((ext_vector_type(8)))  float  v8f;

#if __has_builtin(__builtin_amdgcn_tanhf)
__device__ __forceinline__ float fast_tanh(float x) { return __builtin_amdgcn_tanhf(x); }
#else
__device__ __forceinline__ float fast_tanh(float x) { return tanhf(x); }
#endif

__device__ __forceinline__ float gelu_f(float x) {
    // jax.nn.gelu default (approximate=True, tanh form)
    float x3 = x * x * x;
    return 0.5f * x * (1.0f + fast_tanh(0.7978845608028654f * (x + 0.044715f * x3)));
}

// ---------- conversion kernels ----------

// row-major fp32 -> bf16 (for X)
__global__ void __launch_bounds__(256) cvt_rm_kernel(const float* __restrict__ in,
                                                     __bf16* __restrict__ out, int n) {
    int i = blockIdx.x * blockDim.x + threadIdx.x;
    int stride = gridDim.x * blockDim.x;
    for (; i < n; i += stride) out[i] = (__bf16)in[i];
}

// LDS-tiled transpose-convert: in [E][K][N] f32 -> out [E][N][K] bf16 (B^T pack)
// grid: (N/32, K/32, E), block 256 (32x8)
__global__ void __launch_bounds__(256) cvt_tr_kernel(const float* __restrict__ in,
                                                     __bf16* __restrict__ out,
                                                     int K, int N) {
    __shared__ float tile[32][33];
    const int e  = blockIdx.z;
    const int n0 = blockIdx.x * 32, k0 = blockIdx.y * 32;
    const int tx = threadIdx.x & 31, ty = threadIdx.x >> 5;
    const float* src = in + (size_t)e * K * N;
    __bf16* dst = out + (size_t)e * N * K;
    #pragma unroll
    for (int i = 0; i < 4; ++i)
        tile[ty + i * 8][tx] = src[(size_t)(k0 + ty + i * 8) * N + (n0 + tx)];
    __syncthreads();
    #pragma unroll
    for (int i = 0; i < 4; ++i)
        dst[(size_t)(n0 + ty + i * 8) * K + (k0 + tx)] = (__bf16)tile[tx][ty + i * 8];
}

// ---------- tiled WMMA GEMM: C[M,N] = act(A[M,K] * B[K,N] + bias) ----------
// A : bf16 row-major [NE][CAP][KTOT]
// Bt: bf16 col-major [NE][NTOT][KTOT]  (B transposed -> fragments are contiguous)
// Block tile 128x256x32; 8 waves in 2x4 grid, each wave 64x64 (4x4 WMMA tiles).
// LDS rows padded 64B->80B for bank-conflict-free ds_load_b128.

template<int KTOT, int NTOT, bool DO_GELU>
__global__ void __launch_bounds__(256) moe_gemm_kernel(
    const __bf16* __restrict__ A, const __bf16* __restrict__ Bt,
    const float* __restrict__ bias, __bf16* __restrict__ HOut, float* __restrict__ YOut)
{
    __shared__ __align__(16) char sA[2][128 * 80];   // 10KB/buf: 128 rows x (64B data + 16B pad)
    __shared__ __align__(16) char sB[2][256 * 80];   // 20KB/buf: 256 cols x (64B data + 16B pad)

    const int tid  = threadIdx.x;
    const int wave = tid >> 5, lane = tid & 31;
    const int wr = wave >> 2, wc = wave & 3;          // 2x4 waves over 128x256
    const int lo = (lane >> 4) & 1, ln = lane & 15;
    const int e  = blockIdx.z;
    const int m0 = blockIdx.y * 128;
    const int n0 = blockIdx.x * 256;

    const uint64_t aS = (uint64_t)(A  + (size_t)e * CAP  * KTOT);
    const uint64_t bS = (uint64_t)(Bt + (size_t)e * NTOT * KTOT);
    const uint32_t aL0 = (uint32_t)(uintptr_t)&sA[0][0];
    const uint32_t bL0 = (uint32_t)(uintptr_t)&sB[0][0];

    const int rc = tid >> 2;              // row/col group 0..63
    const int w16 = (tid & 3) * 16;       // 16B chunk within a 64B row/col

    v8f acc[4][4] = {};

    auto issue_stage = [&](int kb, int buf) {
        uint32_t aB = aL0 + (uint32_t)buf * (128 * 80);
        uint32_t bB = bL0 + (uint32_t)buf * (256 * 80);
        #pragma unroll
        for (int i = 0; i < 2; ++i) {                 // A tile: 128 rows x 64B
            int row = rc + i * 64;
            uint32_t voff = (uint32_t)(((m0 + row) * KTOT + kb) * 2 + w16);
            uint32_t lds  = aB + (uint32_t)(row * 80 + w16);
            asm volatile("global_load_async_to_lds_b128 %0, %1, %2"
                         :: "v"(lds), "v"(voff), "s"(aS) : "memory");
        }
        #pragma unroll
        for (int i = 0; i < 4; ++i) {                 // B tile: 256 cols x 64B
            int col = rc + i * 64;
            uint32_t voff = (uint32_t)(((n0 + col) * KTOT + kb) * 2 + w16);
            uint32_t lds  = bB + (uint32_t)(col * 80 + w16);
            asm volatile("global_load_async_to_lds_b128 %0, %1, %2"
                         :: "v"(lds), "v"(voff), "s"(bS) : "memory");
        }
    };

    constexpr int NITER = KTOT / 32;
    issue_stage(0, 0);

    for (int it = 0; it < NITER; ++it) {
        if (it + 1 < NITER) {
            issue_stage((it + 1) * 32, (it + 1) & 1);
            asm volatile("s_wait_asynccnt 6" ::: "memory");   // stage `it` retired
        } else {
            asm volatile("s_wait_asynccnt 0" ::: "memory");
        }
        __syncthreads();

        const char* aBuf = &sA[it & 1][0];
        const char* bBuf = &sB[it & 1][0];

        v16bf af[4], bfr[4];
        #pragma unroll
        for (int mt = 0; mt < 4; ++mt) {
            // A 16x32 layout: lane half `lo` covers K lo*8..+7 and 16+lo*8..+7
            const char* rp = aBuf + (wr * 64 + mt * 16 + ln) * 80 + lo * 16;
            union { uint4 u[2]; v16bf v; } t;
            t.u[0] = *(const uint4*)(rp);
            t.u[1] = *(const uint4*)(rp + 32);
            af[mt] = t.v;
        }
        #pragma unroll
        for (int nt = 0; nt < 4; ++nt) {
            // B 32x16 layout: lane half `lo` covers K lo*8..+7 and 16+lo*8..+7
            const char* cp = bBuf + (wc * 64 + nt * 16 + ln) * 80 + lo * 16;
            union { uint4 u[2]; v16bf v; } t;
            t.u[0] = *(const uint4*)(cp);
            t.u[1] = *(const uint4*)(cp + 32);
            bfr[nt] = t.v;
        }
        #pragma unroll
        for (int mt = 0; mt < 4; ++mt)
            #pragma unroll
            for (int nt = 0; nt < 4; ++nt)
                acc[mt][nt] = __builtin_amdgcn_wmma_f32_16x16x32_bf16(
                    false, af[mt], false, bfr[nt], (short)0, acc[mt][nt], false, false);
        __syncthreads();
    }

    // epilogue: bias (+gelu); C/D layout: M = 8*lo + r within 16-tile, N = ln
    float bv[4];
    #pragma unroll
    for (int nt = 0; nt < 4; ++nt)
        bv[nt] = bias[(size_t)e * NTOT + n0 + wc * 64 + nt * 16 + ln];

    #pragma unroll
    for (int mt = 0; mt < 4; ++mt) {
        #pragma unroll
        for (int nt = 0; nt < 4; ++nt) {
            const int n = n0 + wc * 64 + nt * 16 + ln;
            #pragma unroll
            for (int r = 0; r < 8; ++r) {
                const int m = m0 + wr * 64 + mt * 16 + lo * 8 + r;
                float val = acc[mt][nt][r] + bv[nt];
                if (DO_GELU) {
                    val = gelu_f(val);
                    HOut[((size_t)e * CAP + m) * NTOT + n] = (__bf16)val;
                } else {
                    YOut[((size_t)e * CAP + m) * NTOT + n] = val;
                }
            }
        }
    }
}

// ---------- launch ----------

extern "C" void kernel_launch(void* const* d_in, const int* in_sizes, int n_in,
                              void* d_out, int out_size, void* d_ws, size_t ws_size,
                              hipStream_t stream) {
    const float* X  = (const float*)d_in[0];   // [1, NE*CAP, DM]
    const float* W1 = (const float*)d_in[1];   // [NE, DM, FF]
    const float* b1 = (const float*)d_in[2];   // [NE, FF]
    const float* W2 = (const float*)d_in[3];   // [NE, FF, DM]
    const float* b2 = (const float*)d_in[4];   // [NE, DM]
    float* out = (float*)d_out;                // [1, NE*CAP, DM] fp32

    // workspace: Xb(64MB) | W1t(64MB) | W2t(64MB) | Hb(256MB)
    char* ws = (char*)d_ws;
    __bf16* Xb  = (__bf16*)(ws);
    __bf16* W1t = (__bf16*)(ws + (size_t)NE * CAP * DM * 2);
    __bf16* W2t = (__bf16*)(ws + (size_t)NE * CAP * DM * 2 + (size_t)NE * DM * FF * 2);
    __bf16* Hb  = (__bf16*)(ws + (size_t)NE * CAP * DM * 2 + (size_t)NE * DM * FF * 2
                               + (size_t)NE * FF * DM * 2);

    cvt_rm_kernel<<<2048, 256, 0, stream>>>(X, Xb, NE * CAP * DM);
    // W1 [DM][FF] -> W1t [FF][DM] ; W2 [FF][DM] -> W2t [DM][FF]
    cvt_tr_kernel<<<dim3(FF / 32, DM / 32, NE), 256, 0, stream>>>(W1, W1t, DM, FF);
    cvt_tr_kernel<<<dim3(DM / 32, FF / 32, NE), 256, 0, stream>>>(W2, W2t, FF, DM);

    // GEMM1: H = gelu(X * W1 + b1)   (M=CAP, K=DM, N=FF)
    moe_gemm_kernel<DM, FF, true ><<<dim3(FF / 256, CAP / 128, NE), 256, 0, stream>>>(
        Xb, W1t, b1, Hb, nullptr);
    // GEMM2: Y = H * W2 + b2         (M=CAP, K=FF, N=DM)
    moe_gemm_kernel<FF, DM, false><<<dim3(DM / 256, CAP / 128, NE), 256, 0, stream>>>(
        Hb, W2t, b2, nullptr, out);
}
